// TransformerDecoder_22634477650059
// MI455X (gfx1250) — matrix operand order repaired and checked by reading, not compiled, not run on hardware
//
#include <hip/hip_runtime.h>

// ---------------------------------------------------------------------------
// Problem constants (match reference)
// ---------------------------------------------------------------------------
#define LYR 4
#define BB  2
#define SS  2048
#define DD  2048
#define HH  16
#define KVH 4
#define HD  128
#define FF  8192
#define ROPE_BASE 10000.0f
#define EPSV 1e-6f

#define BSD (BB*SS*DD)          // 8,388,608
#define MROWS (BB*SS)           // 4096

// ---------------------------------------------------------------------------
// bf16 helpers + WMMA fragment types
// ---------------------------------------------------------------------------
typedef __attribute__((ext_vector_type(16))) __bf16 v16bf;
typedef __attribute__((ext_vector_type(8)))  float  v8f;
typedef unsigned int u32x4 __attribute__((ext_vector_type(4)));
typedef int          i32x4 __attribute__((ext_vector_type(4)));
typedef int          i32x8 __attribute__((ext_vector_type(8)));

union Frag {
    v16bf v;
    unsigned int   u[8];
    unsigned short h[16];
};

__device__ __forceinline__ unsigned short f2bf(float f) {
    unsigned int u = __builtin_bit_cast(unsigned int, f);
    unsigned int r = (u + 0x7fffu + ((u >> 16) & 1u)) >> 16;   // RNE
    return (unsigned short)r;
}
__device__ __forceinline__ float bf2f(unsigned short h) {
    unsigned int u = ((unsigned int)h) << 16;
    return __builtin_bit_cast(float, u);
}
__device__ __forceinline__ v8f wmma_bf16(const Frag& a, const Frag& b, v8f c) {
    return __builtin_amdgcn_wmma_f32_16x16x32_bf16(
        /*neg_a=*/false, a.v, /*neg_b=*/false, b.v,
        /*c_mod=*/(short)0, c, /*reuse_a=*/false, /*reuse_b=*/false);
}

// Low 32 bits of a flat shared-memory address == LDS byte offset (aperture map)
__device__ __forceinline__ unsigned lds_off(const void* p) {
    return (unsigned)(unsigned long long)p;
}

// ---------------------------------------------------------------------------
// TDM: issue a 2D bf16 tile load (tileRows x tileCols elems, row stride in
// elems) from global into LDS, padding 1 DWORD after every 32 DWORDs (128B)
// so each padded LDS row is 33 DWORDs (66 bf16).
// ---------------------------------------------------------------------------
__device__ __forceinline__ void tdm_load_tile2d(unsigned ldsAddr, const void* gptr,
                                                unsigned tileCols, unsigned tileRows,
                                                unsigned long long strideElems) {
    unsigned long long ga = (unsigned long long)gptr;
    u32x4 g0;
    g0[0] = 1u;                                            // count=1, user desc
    g0[1] = ldsAddr;                                       // lds_addr
    g0[2] = (unsigned)ga;                                  // global_addr[31:0]
    g0[3] = (unsigned)((ga >> 32) & 0x01FFFFFFu) | 0x80000000u; // addr[56:32], type=2

    const unsigned td0 = 0x40000000u;                      // huge tensor dims: no OOB clip
    const unsigned td1 = 0x40000000u;
    i32x8 g1;
    //        data_size=2B      pad_enable    pad_interval=32DW  pad_amount=1DW
    g1[0] = (int)((1u << 16) | (1u << 20) | (4u << 22) | (0u << 25));
    g1[1] = (int)((td0 & 0xffffu) << 16);                  // abar=0 | tensor_dim0.lo
    g1[2] = (int)((td0 >> 16) | ((td1 & 0xffffu) << 16));  // td0.hi | td1.lo
    g1[3] = (int)((td1 >> 16) | (tileCols << 16));         // td1.hi | tile_dim0
    g1[4] = (int)(tileRows & 0xffffu);                     // tile_dim1 (tile_dim2=0)
    g1[5] = (int)(unsigned)(strideElems & 0xffffffffu);    // tensor_dim0_stride lo
    g1[6] = (int)(unsigned)((strideElems >> 32) & 0xffffu);// stride hi (td1_stride=0)
    g1[7] = 0;

    i32x4 z4 = {0, 0, 0, 0};
#if defined(__clang_major__) && (__clang_major__ >= 23)
    i32x8 z8 = {0, 0, 0, 0, 0, 0, 0, 0};
    __builtin_amdgcn_tensor_load_to_lds(g0, g1, z4, z4, z8, 0);
#else
    __builtin_amdgcn_tensor_load_to_lds(g0, g1, z4, z4, 0);
#endif
}

// ---------------------------------------------------------------------------
// Deterministic whole-tensor sum-of-squares (two pass)
// ---------------------------------------------------------------------------
#define RED_BLOCKS 2048
__global__ __launch_bounds__(256) void k_sumsq_part(const float* __restrict__ x,
                                                    float* __restrict__ part, int n) {
    __shared__ float red[256];
    float s = 0.f;
    for (int i = blockIdx.x * 256 + threadIdx.x; i < n; i += gridDim.x * 256) {
        float v = x[i];
        s += v * v;
    }
    red[threadIdx.x] = s;
    __syncthreads();
    for (int o = 128; o > 0; o >>= 1) {
        if (threadIdx.x < o) red[threadIdx.x] += red[threadIdx.x + o];
        __syncthreads();
    }
    if (threadIdx.x == 0) part[blockIdx.x] = red[0];
}

__global__ __launch_bounds__(256) void k_sumsq_final(const float* __restrict__ part,
                                                     float* __restrict__ out) {
    __shared__ float red[256];
    float s = 0.f;
    for (int i = threadIdx.x; i < RED_BLOCKS; i += 256) s += part[i];
    red[threadIdx.x] = s;
    __syncthreads();
    for (int o = 128; o > 0; o >>= 1) {
        if (threadIdx.x < o) red[threadIdx.x] += red[threadIdx.x + o];
        __syncthreads();
    }
    if (threadIdx.x == 0) out[0] = red[0];
}

__global__ __launch_bounds__(256) void k_rms_scale(const float* __restrict__ h,
                                                   const float* __restrict__ g,
                                                   const float* __restrict__ sumsq,
                                                   unsigned short* __restrict__ a, int n) {
    int i = blockIdx.x * 256 + threadIdx.x;
    if (i >= n) return;
    float inv = 1.f / (sqrtf(sumsq[0]) + EPSV);
    a[i] = f2bf(g[i & (DD - 1)] * h[i] * inv);
}

// ---------------------------------------------------------------------------
// Weight convert + transpose: W (K x N fp32, row-major) -> Wt (N x K bf16)
// 64x64 tiles via LDS.
// ---------------------------------------------------------------------------
__global__ __launch_bounds__(256) void k_wt(const float* __restrict__ W,
                                            unsigned short* __restrict__ Wt,
                                            int K, int N) {
    __shared__ unsigned short t[64][65];
    const int nb = blockIdx.x << 6;
    const int kb = blockIdx.y << 6;
    const int tid = threadIdx.x;
#pragma unroll
    for (int i = 0; i < 16; ++i) {
        int j = i * 256 + tid;
        int r = j >> 6;      // k offset (row in W)
        int c = j & 63;      // n offset (col in W), coalesced
        t[c][r] = f2bf(W[(size_t)(kb + r) * N + nb + c]);
    }
    __syncthreads();
#pragma unroll
    for (int i = 0; i < 16; ++i) {
        int j = i * 256 + tid;
        int r = j >> 6;      // n offset (row in Wt)
        int c = j & 63;      // k offset (col in Wt), coalesced
        Wt[(size_t)(nb + r) * K + kb + c] = t[r][c];
    }
}

// ---------------------------------------------------------------------------
// Tiled bf16 WMMA GEMM with TDM double-buffered staging.
// C(MxN) = A(MxK bf16) @ Bt(NxK bf16)^T + bias
// Block 256 thr (8 waves). Tile 128x128, BK=64 (two WMMA k-steps per stage).
// Wave patch: 32x64 (2x4 WMMA tiles). LDS rows padded to 33 DWORDs by TDM.
// ---------------------------------------------------------------------------
#define GEMM_MODE_STORE 0   // Cf  = acc + bias
#define GEMM_MODE_ADD   1   // Cf += acc + bias      (residual)
#define GEMM_MODE_GATE  2   // Cb  = bf16( U * silu(acc + bias) )
#define GEMM_MODE_BF16  3   // Cb  = bf16( acc + bias )

#define LROW 66             // padded LDS row in bf16 elems (33 DWORDs)

__global__ __launch_bounds__(256) void k_gemm_bf16(
    const unsigned short* __restrict__ A,   // M x K bf16 row-major
    const unsigned short* __restrict__ Bt,  // N x K bf16 row-major (pre-transposed W)
    const float*          __restrict__ bias,// N
    float*                __restrict__ Cf,  // M x N fp32 (modes 0/1)
    unsigned short*       __restrict__ Cb,  // M x N bf16 (modes 2/3)
    const unsigned short* __restrict__ U,   // M x N bf16 (mode 2)
    int M, int N, int K, int mode)
{
    __shared__ __align__(16) unsigned short sA[2][128 * LROW];  // [m][k] padded
    __shared__ __align__(16) unsigned short sB[2][128 * LROW];  // [n][k] padded

    const int tid  = threadIdx.x;
    const int lane = tid & 31;
    const int wid  = tid >> 5;
    const int wm0  = (wid & 3) << 5;               // 0,32,64,96
    const int wn0  = (wid >> 2) << 6;              // 0,64
    const int m0   = blockIdx.y << 7;
    const int n0   = blockIdx.x << 7;

    v8f acc[2][4];
#pragma unroll
    for (int i = 0; i < 2; ++i)
#pragma unroll
        for (int j = 0; j < 4; ++j) acc[i][j] = (v8f)0.f;

    const int kT = K >> 6;
    const unsigned ldsA[2] = { lds_off(&sA[0][0]), lds_off(&sA[1][0]) };
    const unsigned ldsB[2] = { lds_off(&sB[0][0]), lds_off(&sB[1][0]) };

    // ---- prologue: kick off tile 0 via TDM (one wave issues)
    if (wid == 0) {
        tdm_load_tile2d(ldsA[0], A  + (size_t)m0 * K, 64, 128, (unsigned long long)K);
        tdm_load_tile2d(ldsB[0], Bt + (size_t)n0 * K, 64, 128, (unsigned long long)K);
    }

    for (int kb = 0; kb < kT; ++kb) {
        const int cur = kb & 1;
        if (wid == 0) {
            if (kb + 1 < kT) {
                const size_t ko = (size_t)(kb + 1) << 6;
                tdm_load_tile2d(ldsA[cur ^ 1], A  + (size_t)m0 * K + ko, 64, 128,
                                (unsigned long long)K);
                tdm_load_tile2d(ldsB[cur ^ 1], Bt + (size_t)n0 * K + ko, 64, 128,
                                (unsigned long long)K);
                __builtin_amdgcn_s_wait_tensorcnt((short)2);   // tile kb landed
            } else {
                __builtin_amdgcn_s_wait_tensorcnt((short)0);
            }
        }
        __syncthreads();

        const unsigned int* a32 = (const unsigned int*)&sA[cur][0];
        const unsigned short* b16 = &sB[cur][0];

#pragma unroll
        for (int kk = 0; kk < 2; ++kk) {           // two 32-wide k-steps
            Frag aF[2];
#pragma unroll
            for (int i = 0; i < 2; ++i) {
                int mrow = wm0 + i * 16 + (lane & 15);
#pragma unroll
                for (int r = 0; r < 8; ++r) {
                    int u32idx = kk * 16 + ((r < 4) ? 0 : 8) + ((lane < 16) ? 0 : 4) + (r & 3);
                    aF[i].u[r] = a32[mrow * 33 + u32idx];
                }
            }
#pragma unroll
            for (int j = 0; j < 4; ++j) {
                Frag bF;
                int ncol = wn0 + j * 16 + (lane & 15);
#pragma unroll
                for (int r = 0; r < 8; ++r) {
                    int klo = kk * 32 + ((r < 4) ? 0 : 16) + ((lane < 16) ? 0 : 8) + 2 * (r & 3);
                    bF.u[r] = *(const unsigned int*)&b16[ncol * LROW + klo];
                }
#pragma unroll
                for (int i = 0; i < 2; ++i) acc[i][j] = wmma_bf16(aF[i], bF, acc[i][j]);
            }
        }
        __syncthreads();   // all waves done with buf[cur] before TDM overwrites it
    }

    // ---- epilogue (C layout: row = r + 8*(lane>=16), col = lane&15)
#pragma unroll
    for (int i = 0; i < 2; ++i) {
#pragma unroll
        for (int j = 0; j < 4; ++j) {
            int ncol = n0 + wn0 + j * 16 + (lane & 15);
            float bn = bias[ncol];
#pragma unroll
            for (int r = 0; r < 8; ++r) {
                int m = m0 + wm0 + i * 16 + r + ((lane < 16) ? 0 : 8);
                size_t idx = (size_t)m * N + ncol;
                float val = acc[i][j][r] + bn;
                if (mode == GEMM_MODE_STORE) {
                    Cf[idx] = val;
                } else if (mode == GEMM_MODE_ADD) {
                    Cf[idx] += val;
                } else if (mode == GEMM_MODE_GATE) {
                    float uu = bf2f(U[idx]);
                    float sg = val / (1.f + __expf(-val));
                    Cb[idx] = f2bf(uu * sg);
                } else { // GEMM_MODE_BF16
                    Cb[idx] = f2bf(val);
                }
            }
        }
    }
}

// ---------------------------------------------------------------------------
// RoPE in-place on (B, S, nheads, HD) fp32
// ---------------------------------------------------------------------------
__global__ __launch_bounds__(256) void k_rope(float* __restrict__ x, int nheads, int total) {
    int i = blockIdx.x * 256 + threadIdx.x;        // pair index
    if (i >= total) return;
    int j   = i & (HD / 2 - 1);
    int t   = i / (HD / 2);
    int s   = (t / nheads) & (SS - 1);             // token position
    float freq = __powf(ROPE_BASE, -(2.f * (float)j) / (float)HD);
    float ang  = (float)s * freq;
    float c = __cosf(ang), sn = __sinf(ang);
    size_t base = (size_t)t * HD + 2 * j;
    float xe = x[base], xo = x[base + 1];
    x[base]     = xe * c - xo * sn;
    x[base + 1] = xe * sn + xo * c;
}

// ---------------------------------------------------------------------------
// Flash attention (causal, GQA 16q/4kv heads, HD=128)
// Block = 128 thr (4 waves); wave owns 16 q rows; block covers 64 q rows.
// ---------------------------------------------------------------------------
__global__ __launch_bounds__(128) void k_attn(
    const float* __restrict__ q,   // (B,S,H,HD)
    const float* __restrict__ k,   // (B,S,KV,HD)
    const float* __restrict__ vv,  // (B,S,KV,HD)
    unsigned short* __restrict__ o)// (B,S,H,HD) bf16
{
    __shared__ unsigned short sK[64 * 128];        // [key][hd]
    __shared__ unsigned short sV[128 * 64];        // [hd][key]
    __shared__ unsigned short sP[4][16 * 64];      // per-wave P tile [m][k]

    const int tid  = threadIdx.x;
    const int lane = tid & 31;
    const int wid  = tid >> 5;
    const int qt   = blockIdx.x;
    const int h    = blockIdx.y;
    const int b    = blockIdx.z;
    const int kvh  = h >> 2;                       // group = H/KV = 4
    const int q0   = qt * 64 + wid * 16;           // wave's first q row
    const float scale = 0.088388347648318447f;     // 1/sqrt(128)

    Frag qF[4];
    {
        int mrow = q0 + (lane & 15);
        const float* qrow = &q[(((size_t)b * SS + mrow) * HH + h) * HD];
#pragma unroll
        for (int kk = 0; kk < 4; ++kk) {
#pragma unroll
            for (int r = 0; r < 8; ++r) {
                int klo = kk * 32 + ((r < 4) ? 0 : 16) + ((lane < 16) ? 0 : 8) + 2 * (r & 3);
                qF[kk].h[2 * r]     = f2bf(qrow[klo]     * scale);
                qF[kk].h[2 * r + 1] = f2bf(qrow[klo + 1] * scale);
            }
        }
    }

    v8f oacc[8];
#pragma unroll
    for (int t = 0; t < 8; ++t) oacc[t] = (v8f)0.f;
    float mrow_[8], lrow[8];
#pragma unroll
    for (int r = 0; r < 8; ++r) { mrow_[r] = -1e30f; lrow[r] = 0.f; }

    const int nkt = qt + 1;
    for (int kt = 0; kt < nkt; ++kt) {
        __syncthreads();
        for (int idx = tid; idx < 64 * 128; idx += 128) {
            int key = idx >> 7;
            int hd  = idx & 127;
            size_t gi = (((size_t)b * SS + kt * 64 + key) * KVH + kvh) * HD + hd;
            sK[key * 128 + hd] = f2bf(k[gi]);
            sV[hd * 64 + key]  = f2bf(vv[gi]);
        }
        __syncthreads();

        // ---- S = Q @ K^T
        v8f sacc[4];
#pragma unroll
        for (int j = 0; j < 4; ++j) sacc[j] = (v8f)0.f;
#pragma unroll
        for (int j = 0; j < 4; ++j) {
            int key = j * 16 + (lane & 15);
#pragma unroll
            for (int kk = 0; kk < 4; ++kk) {
                Frag kf;
#pragma unroll
                for (int r = 0; r < 8; ++r) {
                    int klo = kk * 32 + ((r < 4) ? 0 : 16) + ((lane < 16) ? 0 : 8) + 2 * (r & 3);
                    kf.u[r] = *(const unsigned int*)&sK[key * 128 + klo];
                }
                sacc[j] = wmma_bf16(qF[kk], kf, sacc[j]);
            }
        }

        // ---- online softmax
#pragma unroll
        for (int r = 0; r < 8; ++r) {
            int row = q0 + r + ((lane < 16) ? 0 : 8);
            float mloc = -3e30f;
#pragma unroll
            for (int j = 0; j < 4; ++j) {
                int col = kt * 64 + j * 16 + (lane & 15);
                float sv = sacc[j][r];
                if (col > row) { sv = -3e30f; sacc[j][r] = sv; }   // causal mask
                mloc = fmaxf(mloc, sv);
            }
            mloc = fmaxf(mloc, __shfl_xor(mloc, 1, 32));
            mloc = fmaxf(mloc, __shfl_xor(mloc, 2, 32));
            mloc = fmaxf(mloc, __shfl_xor(mloc, 4, 32));
            mloc = fmaxf(mloc, __shfl_xor(mloc, 8, 32));
            float mnew  = fmaxf(mrow_[r], mloc);
            float alpha = __expf(mrow_[r] - mnew);
            mrow_[r] = mnew;
            float psum = 0.f;
#pragma unroll
            for (int j = 0; j < 4; ++j) {
                float p = __expf(sacc[j][r] - mnew);
                sacc[j][r] = p;
                psum += p;
            }
            psum += __shfl_xor(psum, 1, 32);
            psum += __shfl_xor(psum, 2, 32);
            psum += __shfl_xor(psum, 4, 32);
            psum += __shfl_xor(psum, 8, 32);
            lrow[r] = lrow[r] * alpha + psum;
#pragma unroll
            for (int t = 0; t < 8; ++t) oacc[t][r] *= alpha;
        }

        // ---- P -> per-wave LDS then A-fragment reload
#pragma unroll
        for (int j = 0; j < 4; ++j)
#pragma unroll
            for (int r = 0; r < 8; ++r) {
                int prow = r + ((lane < 16) ? 0 : 8);
                sP[wid][prow * 64 + j * 16 + (lane & 15)] = f2bf(sacc[j][r]);
            }
        asm volatile("s_wait_dscnt 0x0" ::: "memory");

        Frag pF[2];
#pragma unroll
        for (int kk = 0; kk < 2; ++kk) {
            int prow = lane & 15;
#pragma unroll
            for (int r = 0; r < 8; ++r) {
                int klo = kk * 32 + ((r < 4) ? 0 : 16) + ((lane < 16) ? 0 : 8) + 2 * (r & 3);
                pF[kk].u[r] = *(const unsigned int*)&sP[wid][prow * 64 + klo];
            }
        }
        // ---- O += P @ V
#pragma unroll
        for (int t = 0; t < 8; ++t) {
            int hd = t * 16 + (lane & 15);
#pragma unroll
            for (int kk = 0; kk < 2; ++kk) {
                Frag vf;
#pragma unroll
                for (int r = 0; r < 8; ++r) {
                    int klo = kk * 32 + ((r < 4) ? 0 : 16) + ((lane < 16) ? 0 : 8) + 2 * (r & 3);
                    vf.u[r] = *(const unsigned int*)&sV[hd * 64 + klo];
                }
                oacc[t] = wmma_bf16(pF[kk], vf, oacc[t]);
            }
        }
    }

    // ---- normalize and store bf16
#pragma unroll
    for (int t = 0; t < 8; ++t) {
        int hd = t * 16 + (lane & 15);
#pragma unroll
        for (int r = 0; r < 8; ++r) {
            int row = q0 + r + ((lane < 16) ? 0 : 8);
            float val = oacc[t][r] / lrow[r];
            o[(((size_t)b * SS + row) * HH + h) * HD + hd] = f2bf(val);
        }
    }
}

// ---------------------------------------------------------------------------
// Host-side orchestration
// ---------------------------------------------------------------------------
extern "C" void kernel_launch(void* const* d_in, const int* in_sizes, int n_in,
                              void* d_out, int out_size, void* d_ws, size_t ws_size,
                              hipStream_t stream) {
    (void)in_sizes; (void)n_in; (void)out_size; (void)ws_size;
    const float* x      = (const float*)d_in[0];
    const float* Wq     = (const float*)d_in[1];
    const float* bq     = (const float*)d_in[2];
    const float* Wk     = (const float*)d_in[3];
    const float* bk     = (const float*)d_in[4];
    const float* Wv     = (const float*)d_in[5];
    const float* bv     = (const float*)d_in[6];
    const float* Wo     = (const float*)d_in[7];
    const float* bo     = (const float*)d_in[8];
    const float* W0     = (const float*)d_in[9];
    const float* b0     = (const float*)d_in[10];
    const float* W1     = (const float*)d_in[11];
    const float* b1     = (const float*)d_in[12];
    const float* W2     = (const float*)d_in[13];
    const float* b2     = (const float*)d_in[14];
    const float* g_attn = (const float*)d_in[15];
    const float* g_ffn  = (const float*)d_in[16];

    // ---- workspace carve-out
    unsigned char* p = (unsigned char*)d_ws;
    float*          hbuf = (float*)p;          p += (size_t)BSD * 4;             // 32 MB
    unsigned short* abuf = (unsigned short*)p; p += (size_t)BSD * 2;             // 16 MB
    float*          qb   = (float*)p;          p += (size_t)MROWS * HH * HD * 4; // 32 MB
    float*          kb   = (float*)p;          p += (size_t)MROWS * KVH * HD * 4;// 8 MB
    float*          vb   = (float*)p;          p += (size_t)MROWS * KVH * HD * 4;// 8 MB
    unsigned short* ubuf = (unsigned short*)p; p += (size_t)MROWS * FF * 2;      // 64 MB
    unsigned short* tbuf = (unsigned short*)p; p += (size_t)MROWS * FF * 2;      // 64 MB
    // per-layer transposed bf16 weights (reused each layer), 116 MB
    unsigned short* wtQ  = (unsigned short*)p; p += (size_t)(HH * HD) * DD * 2;
    unsigned short* wtK  = (unsigned short*)p; p += (size_t)(KVH * HD) * DD * 2;
    unsigned short* wtV  = (unsigned short*)p; p += (size_t)(KVH * HD) * DD * 2;
    unsigned short* wtO  = (unsigned short*)p; p += (size_t)DD * DD * 2;
    unsigned short* wt0  = (unsigned short*)p; p += (size_t)FF * DD * 2;
    unsigned short* wt1  = (unsigned short*)p; p += (size_t)FF * DD * 2;
    unsigned short* wt2  = (unsigned short*)p; p += (size_t)DD * FF * 2;
    float*          part = (float*)p;          p += RED_BLOCKS * 4;
    float*          nrm  = (float*)p;          p += 256;

    hipMemcpyAsync(hbuf, x, (size_t)BSD * 4, hipMemcpyDeviceToDevice, stream);

    const dim3 gemmBlk(256);
    const int ropeQ = MROWS * HH * (HD / 2);
    const int ropeK = MROWS * KVH * (HD / 2);

    for (int i = 0; i < LYR; ++i) {
        const float* Wq_i = Wq + (size_t)i * DD * (HH * HD);
        const float* Wk_i = Wk + (size_t)i * DD * (KVH * HD);
        const float* Wv_i = Wv + (size_t)i * DD * (KVH * HD);
        const float* Wo_i = Wo + (size_t)i * DD * DD;
        const float* W0_i = W0 + (size_t)i * DD * FF;
        const float* W1_i = W1 + (size_t)i * DD * FF;
        const float* W2_i = W2 + (size_t)i * FF * DD;

        // ---- convert + transpose this layer's weights to bf16 [N][K]
        k_wt<<<dim3((HH * HD) / 64,  DD / 64), 256, 0, stream>>>(Wq_i, wtQ, DD, HH * HD);
        k_wt<<<dim3((KVH * HD) / 64, DD / 64), 256, 0, stream>>>(Wk_i, wtK, DD, KVH * HD);
        k_wt<<<dim3((KVH * HD) / 64, DD / 64), 256, 0, stream>>>(Wv_i, wtV, DD, KVH * HD);
        k_wt<<<dim3(DD / 64,         DD / 64), 256, 0, stream>>>(Wo_i, wtO, DD, DD);
        k_wt<<<dim3(FF / 64,         DD / 64), 256, 0, stream>>>(W0_i, wt0, DD, FF);
        k_wt<<<dim3(FF / 64,         DD / 64), 256, 0, stream>>>(W1_i, wt1, DD, FF);
        k_wt<<<dim3(DD / 64,         FF / 64), 256, 0, stream>>>(W2_i, wt2, FF, DD);

        // ---- attention RMS norm (whole-tensor Frobenius norm)
        k_sumsq_part<<<RED_BLOCKS, 256, 0, stream>>>(hbuf, part, BSD);
        k_sumsq_final<<<1, 256, 0, stream>>>(part, nrm);
        k_rms_scale<<<BSD / 256, 256, 0, stream>>>(hbuf, g_attn + (size_t)i * DD, nrm, abuf, BSD);

        // ---- QKV projections
        k_gemm_bf16<<<dim3((HH * HD) / 128, MROWS / 128), gemmBlk, 0, stream>>>(
            abuf, wtQ, bq + (size_t)i * HH * HD, qb, nullptr, nullptr,
            MROWS, HH * HD, DD, GEMM_MODE_STORE);
        k_gemm_bf16<<<dim3((KVH * HD) / 128, MROWS / 128), gemmBlk, 0, stream>>>(
            abuf, wtK, bk + (size_t)i * KVH * HD, kb, nullptr, nullptr,
            MROWS, KVH * HD, DD, GEMM_MODE_STORE);
        k_gemm_bf16<<<dim3((KVH * HD) / 128, MROWS / 128), gemmBlk, 0, stream>>>(
            abuf, wtV, bv + (size_t)i * KVH * HD, vb, nullptr, nullptr,
            MROWS, KVH * HD, DD, GEMM_MODE_STORE);

        // ---- RoPE
        k_rope<<<(ropeQ + 255) / 256, 256, 0, stream>>>(qb, HH, ropeQ);
        k_rope<<<(ropeK + 255) / 256, 256, 0, stream>>>(kb, KVH, ropeK);

        // ---- flash attention -> abuf (bf16)
        k_attn<<<dim3(SS / 64, HH, BB), 128, 0, stream>>>(qb, kb, vb, abuf);

        // ---- output projection + residual: h += o @ Wo + bo
        k_gemm_bf16<<<dim3(DD / 128, MROWS / 128), gemmBlk, 0, stream>>>(
            abuf, wtO, bo + (size_t)i * DD, hbuf, nullptr, nullptr,
            MROWS, DD, DD, GEMM_MODE_ADD);

        // ---- FFN RMS norm
        k_sumsq_part<<<RED_BLOCKS, 256, 0, stream>>>(hbuf, part, BSD);
        k_sumsq_final<<<1, 256, 0, stream>>>(part, nrm);
        k_rms_scale<<<BSD / 256, 256, 0, stream>>>(hbuf, g_ffn + (size_t)i * DD, nrm, abuf, BSD);

        // ---- u = f@W0 + b0 (bf16); t = u * silu(f@W1 + b1) (bf16)
        k_gemm_bf16<<<dim3(FF / 128, MROWS / 128), gemmBlk, 0, stream>>>(
            abuf, wt0, b0 + (size_t)i * FF, nullptr, ubuf, nullptr,
            MROWS, FF, DD, GEMM_MODE_BF16);
        k_gemm_bf16<<<dim3(FF / 128, MROWS / 128), gemmBlk, 0, stream>>>(
            abuf, wt1, b1 + (size_t)i * FF, nullptr, tbuf, ubuf,
            MROWS, FF, DD, GEMM_MODE_GATE);

        // ---- h += t @ W2 + b2
        k_gemm_bf16<<<dim3(DD / 128, MROWS / 128), gemmBlk, 0, stream>>>(
            tbuf, wt2, b2 + (size_t)i * DD, hbuf, nullptr, nullptr,
            MROWS, DD, FF, GEMM_MODE_ADD);
    }

    hipMemcpyAsync(d_out, hbuf, (size_t)BSD * 4, hipMemcpyDeviceToDevice, stream);
}
